// MultiLateInteractiveRetriever_10754598109472
// MI455X (gfx1250) — matrix operand order repaired and checked by reading, not compiled, not run on hardware
//
#include <hip/hip_runtime.h>

// ---------------------------------------------------------------------------
// MultiLateInteractiveRetriever for MI455X (gfx1250, wave32, WMMA).
// All dense GEMMs run on v_wmma_f32_16x16x32_bf16 with *bf16-resident*
// operands: weights/inputs converted once, intermediates stored bf16 by the
// producing kernel, so the GEMM inner loop is pure b128 loads + WMMA.
// Cross-lane reductions use explicit ds_swizzle_b32 (no index clamp math).
// ---------------------------------------------------------------------------

#define TOKS   1024      // B*T
#define DMODEL 256
#define FFDIM  2048
#define NHEAD  4
#define DHEAD  64
#define NCTX   2000
#define JSW    16
#define JPH    12
#define KSWTOK (NCTX*JSW) // 32000
#define KPHTOK (NCTX*JPH) // 24000

typedef __attribute__((ext_vector_type(16))) __bf16 v16bf;
typedef __attribute__((ext_vector_type(8)))  __bf16 v8bf;
typedef __attribute__((ext_vector_type(8)))  float  v8f;
typedef __attribute__((ext_vector_type(4)))  float  vf4;

static __device__ __forceinline__ float fast_sigmoid(float x) {
  return __builtin_amdgcn_rcpf(1.0f + __expf(-x));   // v_exp + v_rcp
}

// ds_swizzle_b32 group-of-32 xor pattern: imm = (xor_mask<<5) | and_mask(0x1f)
template <int XM>
static __device__ __forceinline__ float swz_xor(float x) {
  int y = __builtin_amdgcn_ds_swizzle(__builtin_bit_cast(int, x),
                                      (XM << 5) | 0x1f);
  return __builtin_bit_cast(float, y);
}
template <int XM>
static __device__ __forceinline__ float xor_max(float x) {
  return fmaxf(x, swz_xor<XM>(x));
}
template <int XM>
static __device__ __forceinline__ float xor_add(float x) {
  return x + swz_xor<XM>(x);
}

// A fragment (16x32 bf16, MxK): lane<16 holds K = {k0..k0+15},
// lane>=16 holds K = {k0+16..k0+31} (same k order as B fragment).
static __device__ __forceinline__ v16bf frag_a_bf(const __bf16* __restrict__ row,
                                                  int k0, int hi) {
  return *(const v16bf*)(row + k0 + (hi ? 16 : 0));
}

// B fragment (32x16 bf16, KxN): column n = lane&15 is a K-contiguous row;
// lane<16 holds K = {k0..k0+15}, lane>=16 holds K = {k0+16..k0+31}.
static __device__ __forceinline__ v16bf frag_b_bf(const __bf16* __restrict__ row,
                                                  int k0, int hi) {
  return *(const v16bf*)(row + k0 + (hi ? 16 : 0));  // 32B -> 2x b128
}

static __device__ __forceinline__ v8f wmma_bf16(v16bf a, v16bf b, v8f c) {
  return __builtin_amdgcn_wmma_f32_16x16x32_bf16(false, a, false, b, (short)0, c,
                                                 false, false);
}

// f32 -> bf16 bulk convert, 8 elements / lane, b128 in / b128 out
__global__ __launch_bounds__(256)
void cvt_bf16_k(const float* __restrict__ in, __bf16* __restrict__ out, int n8) {
  int i = blockIdx.x * 256 + threadIdx.x;
  if (i >= n8) return;
  const vf4* p = (const vf4*)(in + (size_t)i * 8);
  vf4 a = p[0], b = p[1];
  v8bf o;
#pragma unroll
  for (int k = 0; k < 4; ++k) { o[k] = (__bf16)a[k]; o[4 + k] = (__bf16)b[k]; }
  *(v8bf*)(out + (size_t)i * 8) = o;
}

// ---------------------------------------------------------------------------
// Generic batched GEMM:  O[m,n] = epilogue( sum_k A[m,k] * B[n,k] ), bf16 in.
// Batch z = (b = z/Hs, h = z%Hs) with per-tensor b/h strides (attention views).
// epilogue: t = acc + bias[n]; if act: t = swish(t); t *= scale;
//           if HAS_R: t += R[m,n] (f32);  store f32 or bf16 per OBF16.
// One wave computes one 16x16 tile; 8 waves / 256-thread block.
// ---------------------------------------------------------------------------
template <bool HAS_R, bool OBF16>
__global__ __launch_bounds__(256)
void gemm_wmma_k(const __bf16* __restrict__ A, long sAb, long sAh, int lda,
                 const __bf16* __restrict__ B, long sBb, long sBh, int ldb,
                 const float* __restrict__ bias,
                 const float* __restrict__ R, long sRb, long sRh, int ldr,
                 void* __restrict__ Ov, long sOb, long sOh, int ldo,
                 int N, int K, int Hs, float scale, int act) {
  int z = blockIdx.z;
  int zb = z / Hs, zh = z % Hs;
  A += (size_t)zb * sAb + (size_t)zh * sAh;
  B += (size_t)zb * sBb + (size_t)zh * sBh;
  if (HAS_R) R += (size_t)zb * sRb + (size_t)zh * sRh;

  int wave = threadIdx.x >> 5;
  int lane = threadIdx.x & 31;
  int tn = blockIdx.y * 8 + wave;
  if (tn * 16 >= N) return;          // wave-uniform; EXEC stays all-ones
  int tm = blockIdx.x;
  int l16 = lane & 15;
  int hi  = lane >> 4;

  const __bf16* arow = A + (size_t)(tm * 16 + l16) * lda;
  const __bf16* brow = B + (size_t)(tn * 16 + l16) * ldb;

  v8f acc = {};
#pragma unroll 2
  for (int k0 = 0; k0 < K; k0 += 32) {
    if (k0 + 32 < K) {
      __builtin_prefetch(arow + k0 + 32, 0, 3);   // global_prefetch_b8
      __builtin_prefetch(brow + k0 + 32, 0, 3);
    }
    v16bf af = frag_a_bf(arow, k0, hi);
    v16bf bf = frag_b_bf(brow, k0, hi);
    acc = wmma_bf16(af, bf, acc);
  }

  int row0 = tm * 16 + hi * 8;       // C layout: VGPR r -> row r / r+8
  int col  = tn * 16 + l16;
  float bv = bias ? bias[col] : 0.0f;
  float t[8];
#pragma unroll
  for (int r = 0; r < 8; ++r) t[r] = acc[r] + bv;
  if (act) {
#pragma unroll
    for (int r = 0; r < 8; ++r) t[r] *= fast_sigmoid(t[r]);
  }
#pragma unroll
  for (int r = 0; r < 8; ++r) t[r] *= scale;
  if (HAS_R) {
#pragma unroll
    for (int r = 0; r < 8; ++r) t[r] += R[(size_t)(row0 + r) * ldr + col];
  }
  if (OBF16) {
    __bf16* O = (__bf16*)Ov + (size_t)zb * sOb + (size_t)zh * sOh;
#pragma unroll
    for (int r = 0; r < 8; ++r) O[(size_t)(row0 + r) * ldo + col] = (__bf16)t[r];
  } else {
    float* O = (float*)Ov + (size_t)zb * sOb + (size_t)zh * sOh;
#pragma unroll
    for (int r = 0; r < 8; ++r) O[(size_t)(row0 + r) * ldo + col] = t[r];
  }
}

// ---------------------------------------------------------------------------
// LayerNorm over D=256 (one wave / token, 8 floats / lane), optional post-norm
// f32 residual add; emits bf16 (consumers are WMMA GEMMs / similarity).
// ---------------------------------------------------------------------------
__global__ __launch_bounds__(256)
void layernorm_k(const float* __restrict__ x, const float* __restrict__ g,
                 const float* __restrict__ beta, const float* __restrict__ addres,
                 __bf16* __restrict__ out, int ntok) {
  int wave = threadIdx.x >> 5;
  int lane = threadIdx.x & 31;
  int tok  = blockIdx.x * 8 + wave;
  if (tok >= ntok) return;
  const float* xr = x + (size_t)tok * DMODEL + lane * 8;
  float v[8];
  float s = 0.0f, sq = 0.0f;
#pragma unroll
  for (int i = 0; i < 8; ++i) { v[i] = xr[i]; s += v[i]; sq += v[i] * v[i]; }
  s = xor_add<16>(s); s = xor_add<8>(s); s = xor_add<4>(s);
  s = xor_add<2>(s);  s = xor_add<1>(s);
  sq = xor_add<16>(sq); sq = xor_add<8>(sq); sq = xor_add<4>(sq);
  sq = xor_add<2>(sq);  sq = xor_add<1>(sq);
  float mean = s * (1.0f / DMODEL);
  float var  = sq * (1.0f / DMODEL) - mean * mean;
  float inv  = rsqrtf(var + 1e-5f);
  const float* ar = addres ? addres + (size_t)tok * DMODEL + lane * 8 : nullptr;
  v8bf o;
#pragma unroll
  for (int i = 0; i < 8; ++i) {
    int c = lane * 8 + i;
    float t = (v[i] - mean) * inv * g[c] + beta[c];
    if (ar) t += ar[i];
    o[i] = (__bf16)t;
  }
  *(v8bf*)(out + (size_t)tok * DMODEL + lane * 8) = o;
}

// GLU: out[m,c] = a[m,c] * sigmoid(a[m,c+256]) on (TOKS,512) -> (TOKS,256) f32
__global__ __launch_bounds__(256)
void glu_k(const float* __restrict__ x, float* __restrict__ out) {
  int i = blockIdx.x * 256 + threadIdx.x;   // < TOKS*256
  int c = i & 255, m = i >> 8;
  float a = x[(size_t)m * 512 + c];
  float b = x[(size_t)m * 512 + 256 + c];
  out[i] = a * fast_sigmoid(b);
}

// Depthwise conv (K=31, same pad) + eval-BN (mean0,var1) + swish -> bf16
__global__ __launch_bounds__(256)
void dwconv_bn_swish_k(const float* __restrict__ x, const float* __restrict__ w,
                       const float* __restrict__ wb, const float* __restrict__ g,
                       const float* __restrict__ bb, __bf16* __restrict__ out) {
  int i = blockIdx.x * 256 + threadIdx.x;   // < 4*256*256
  int c = i & 255, t = (i >> 8) & 255, b = i >> 16;
  const float* xb = x + (size_t)b * 65536;
  float acc = wb[c];
#pragma unroll
  for (int k = 0; k < 31; ++k) {
    int tt = t + k - 15;
    if (tt >= 0 && tt < 256) acc += xb[(size_t)tt * 256 + c] * w[k * 256 + c];
  }
  float y = g[c] * acc * rsqrtf(1.0f + 1e-5f) + bb[c];
  out[i] = (__bf16)(y * fast_sigmoid(y));
}

// V transpose (bf16): vt[b,h,d,t] = v[b,t,h*64+d]  (makes attn*V K-contiguous)
__global__ __launch_bounds__(256)
void vtrans_k(const __bf16* __restrict__ v, __bf16* __restrict__ vt) {
  int i = blockIdx.x * 256 + threadIdx.x;   // < 262144
  int t = i & 255, d = (i >> 8) & 63, h = (i >> 14) & 3, b = i >> 16;
  vt[i] = v[(size_t)b * 65536 + (size_t)t * 256 + h * 64 + d];
}

// Softmax over rows of length 256 (attention scores), f32 in -> bf16 out
__global__ __launch_bounds__(256)
void softmax256_k(const float* __restrict__ in, __bf16* __restrict__ out) {
  __shared__ float red[256];
  int tid = threadIdx.x;
  const float* p = in + (size_t)blockIdx.x * 256;
  float v = p[tid];
  red[tid] = v; __syncthreads();
  for (int s = 128; s >= 1; s >>= 1) {
    if (tid < s) red[tid] = fmaxf(red[tid], red[tid + s]);
    __syncthreads();
  }
  float m = red[0]; __syncthreads();
  float e = __expf(v - m);
  red[tid] = e; __syncthreads();
  for (int s = 128; s >= 1; s >>= 1) {
    if (tid < s) red[tid] += red[tid + s];
    __syncthreads();
  }
  out[(size_t)blockIdx.x * 256 + tid] = (__bf16)(e * __builtin_amdgcn_rcpf(red[0]));
}

// ---------------------------------------------------------------------------
// Fused ColBERT similarity:
//   maxsim[t,c] = max_j(q_sw[t]·k_sw[c,j]) + max_j(q_ph[t]·k_ph[c,j])
// One wave: 16 tokens x 1 context.  Two 16x16x256 bf16 WMMA chains, then
// row-max across the 16 tile columns (J) via ds_swizzle xor-swaps 8/4/2/1
// (stays within each 16-lane half).  ph tile columns 12..15 read a duplicate
// of key row 0 (always in-bounds); those lanes are masked to -inf in the
// J-max, so no B-fragment padding is needed.
// ---------------------------------------------------------------------------
__global__ __launch_bounds__(256)
void similarity_k(const __bf16* __restrict__ qsw, const __bf16* __restrict__ ksw,
                  const __bf16* __restrict__ qph, const __bf16* __restrict__ kph,
                  float* __restrict__ maxsim) {
  int wave = threadIdx.x >> 5;
  int lane = threadIdx.x & 31;
  int c  = blockIdx.y * 8 + wave;          // 0..1999
  int tm = blockIdx.x;                     // 0..63 token tile
  int l16 = lane & 15;
  int hi  = lane >> 4;

  const __bf16* qsr = qsw + (size_t)(tm * 16 + l16) * DMODEL;
  const __bf16* qpr = qph + (size_t)(tm * 16 + l16) * DMODEL;
  const __bf16* ksr = ksw + ((size_t)c * JSW + l16) * DMODEL;
  int jph = (l16 < JPH) ? l16 : 0;         // in-bounds duplicate for pad lanes
  const __bf16* kpr = kph + ((size_t)c * JPH + jph) * DMODEL;

  v8f asw = {}, aph = {};
#pragma unroll 2
  for (int k0 = 0; k0 < DMODEL; k0 += 32) {
    v16bf a0 = frag_a_bf(qsr, k0, hi);
    v16bf b0 = frag_b_bf(ksr, k0, hi);
    asw = wmma_bf16(a0, b0, asw);
    v16bf a1 = frag_a_bf(qpr, k0, hi);
    v16bf b1 = frag_b_bf(kpr, k0, hi);
    aph = wmma_bf16(a1, b1, aph);
  }

#pragma unroll
  for (int r = 0; r < 8; ++r) {
    float s = asw[r];
    float p = (l16 < JPH) ? aph[r] : -3.4e38f;
    s = xor_max<8>(s); s = xor_max<4>(s); s = xor_max<2>(s); s = xor_max<1>(s);
    p = xor_max<8>(p); p = xor_max<4>(p); p = xor_max<2>(p); p = xor_max<1>(p);
    if (l16 == 0) {
      int row = tm * 16 + hi * 8 + r;
      maxsim[(size_t)row * NCTX + c] = s + p;
    }
  }
}

// Final softmax over C=2000 per (b,t) row
__global__ __launch_bounds__(256)
void softmax2000_k(const float* __restrict__ ms, float* __restrict__ out) {
  __shared__ float red[256];
  int tid = threadIdx.x;
  const float* p = ms + (size_t)blockIdx.x * NCTX;
  float m = -3.4e38f;
  for (int c = tid; c < NCTX; c += 256) m = fmaxf(m, p[c]);
  red[tid] = m; __syncthreads();
  for (int s = 128; s >= 1; s >>= 1) {
    if (tid < s) red[tid] = fmaxf(red[tid], red[tid + s]);
    __syncthreads();
  }
  m = red[0]; __syncthreads();
  float sum = 0.0f;
  for (int c = tid; c < NCTX; c += 256) sum += __expf(p[c] - m);
  red[tid] = sum; __syncthreads();
  for (int s = 128; s >= 1; s >>= 1) {
    if (tid < s) red[tid] += red[tid + s];
    __syncthreads();
  }
  float inv = __builtin_amdgcn_rcpf(red[0]);
  float* o = out + (size_t)blockIdx.x * NCTX;
  for (int c = tid; c < NCTX; c += 256) o[c] = __expf(p[c] - m) * inv;
}

// ---------------------------------------------------------------------------
// Host orchestration
// ---------------------------------------------------------------------------
static void launch_gemm(hipStream_t s,
                        const __bf16* A, long sAb, long sAh, int lda,
                        const __bf16* B, long sBb, long sBh, int ldb,
                        const float* bias,
                        const float* R, long sRb, long sRh, int ldr,
                        void* O, long sOb, long sOh, int ldo, bool obf16,
                        int M, int N, int K, int Hs, int nZ, float scale, int act) {
  dim3 grid(M / 16, (N + 127) / 128, nZ);
  if (R) {
    if (obf16)
      gemm_wmma_k<true, true><<<grid, 256, 0, s>>>(A, sAb, sAh, lda, B, sBb, sBh, ldb,
          bias, R, sRb, sRh, ldr, O, sOb, sOh, ldo, N, K, Hs, scale, act);
    else
      gemm_wmma_k<true, false><<<grid, 256, 0, s>>>(A, sAb, sAh, lda, B, sBb, sBh, ldb,
          bias, R, sRb, sRh, ldr, O, sOb, sOh, ldo, N, K, Hs, scale, act);
  } else {
    if (obf16)
      gemm_wmma_k<false, true><<<grid, 256, 0, s>>>(A, sAb, sAh, lda, B, sBb, sBh, ldb,
          bias, nullptr, 0, 0, 0, O, sOb, sOh, ldo, N, K, Hs, scale, act);
    else
      gemm_wmma_k<false, false><<<grid, 256, 0, s>>>(A, sAb, sAh, lda, B, sBb, sBh, ldb,
          bias, nullptr, 0, 0, 0, O, sOb, sOh, ldo, N, K, Hs, scale, act);
  }
}

static void launch_cvt(hipStream_t s, const float* in, __bf16* out, size_t n) {
  int n8 = (int)(n / 8);
  cvt_bf16_k<<<(n8 + 255) / 256, 256, 0, s>>>(in, out, n8);
}

extern "C" void kernel_launch(void* const* d_in, const int* in_sizes, int n_in,
                              void* d_out, int out_size, void* d_ws, size_t ws_size,
                              hipStream_t stream) {
  (void)in_sizes; (void)n_in; (void)out_size; (void)ws_size;

  const float* model    = (const float*)d_in[0];   // (4,256,256)
  const float* ctx      = (const float*)d_in[1];   // (2000,16,256)
  const float* xph      = (const float*)d_in[2];   // (2000,12,256)
  const float* ln_ffm_g = (const float*)d_in[3];
  const float* ln_ffm_b = (const float*)d_in[4];
  const float* ln_mha_g = (const float*)d_in[5];
  const float* ln_mha_b = (const float*)d_in[6];
  const float* ln_conv_g= (const float*)d_in[7];
  const float* ln_conv_b= (const float*)d_in[8];
  const float* ln_ff_g  = (const float*)d_in[9];
  const float* ln_ff_b  = (const float*)d_in[10];
  const float* ln_fin_g = (const float*)d_in[11];
  const float* ln_fin_b = (const float*)d_in[12];
  const float* ffm_w1   = (const float*)d_in[13];  // (2048,256)
  const float* ffm_b1   = (const float*)d_in[14];
  const float* ffm_w2   = (const float*)d_in[15];  // (256,2048)
  const float* ffm_b2   = (const float*)d_in[16];
  const float* ff_w1    = (const float*)d_in[17];
  const float* ff_b1    = (const float*)d_in[18];
  const float* ff_w2    = (const float*)d_in[19];
  const float* ff_b2    = (const float*)d_in[20];
  const float* wq       = (const float*)d_in[21];
  const float* wq_b     = (const float*)d_in[22];
  const float* wk       = (const float*)d_in[23];
  const float* wk_b     = (const float*)d_in[24];
  const float* wv       = (const float*)d_in[25];
  const float* wv_b     = (const float*)d_in[26];
  const float* wo       = (const float*)d_in[27];
  const float* wo_b     = (const float*)d_in[28];
  const float* pw1_w    = (const float*)d_in[29];  // (512,256)
  const float* pw1_b    = (const float*)d_in[30];
  const float* dw_w     = (const float*)d_in[31];  // (31,1,256)
  const float* dw_b     = (const float*)d_in[32];
  const float* bn_g     = (const float*)d_in[33];
  const float* bn_b     = (const float*)d_in[34];
  const float* pw2_w    = (const float*)d_in[35];
  const float* pw2_b    = (const float*)d_in[36];
  const float* ph_q_w   = (const float*)d_in[37];
  const float* ph_q_b   = (const float*)d_in[38];
  const float* sw_k_w   = (const float*)d_in[39];
  const float* sw_k_b   = (const float*)d_in[40];
  const float* ph_k_w   = (const float*)d_in[41];
  const float* ph_k_b   = (const float*)d_in[42];

  // ---- workspace: f32 region then bf16 region ----
  float* ws = (float*)d_ws;
  size_t off = 0;
  float* X     = ws + off; off += (size_t)TOKS * DMODEL;   // residual state
  float* PW1B  = ws + off; off += (size_t)TOKS * 512;      // pw1 out
  float* GLUB  = ws + off; off += (size_t)TOKS * DMODEL;   // glu out
  float* ATTN  = ws + off; off += (size_t)16 * 256 * 256;  // raw scores
  float* MAXS  = ws + off; off += (size_t)TOKS * NCTX;     // maxsim

  __bf16* hb = (__bf16*)(ws + off);
  size_t ho = 0;
  __bf16* MODELH = hb + ho; ho += (size_t)TOKS * DMODEL;
  __bf16* CTXH   = hb + ho; ho += (size_t)KSWTOK * DMODEL;
  __bf16* XPHH   = hb + ho; ho += (size_t)KPHTOK * DMODEL;
  __bf16* LNBH   = hb + ho; ho += (size_t)TOKS * DMODEL;
  __bf16* FFBH   = hb + ho; ho += (size_t)TOKS * FFDIM;
  __bf16* QH     = hb + ho; ho += (size_t)TOKS * DMODEL;
  __bf16* KH     = hb + ho; ho += (size_t)TOKS * DMODEL;
  __bf16* VH     = hb + ho; ho += (size_t)TOKS * DMODEL;
  __bf16* VTH    = hb + ho; ho += (size_t)TOKS * DMODEL;
  __bf16* ATTNH  = hb + ho; ho += (size_t)16 * 256 * 256;
  __bf16* OBH    = hb + ho; ho += (size_t)TOKS * DMODEL;
  __bf16* CONVH  = hb + ho; ho += (size_t)TOKS * DMODEL;
  __bf16* QSWH   = hb + ho; ho += (size_t)TOKS * DMODEL;
  __bf16* QPHH   = hb + ho; ho += (size_t)TOKS * DMODEL;
  __bf16* KEYSWH = hb + ho; ho += (size_t)KSWTOK * DMODEL;
  __bf16* KEYPHH = hb + ho; ho += (size_t)KPHTOK * DMODEL;
  __bf16* WFFM1H = hb + ho; ho += (size_t)FFDIM * DMODEL;
  __bf16* WFFM2H = hb + ho; ho += (size_t)DMODEL * FFDIM;
  __bf16* WFF1H  = hb + ho; ho += (size_t)FFDIM * DMODEL;
  __bf16* WFF2H  = hb + ho; ho += (size_t)DMODEL * FFDIM;
  __bf16* WQH    = hb + ho; ho += (size_t)DMODEL * DMODEL;
  __bf16* WKH    = hb + ho; ho += (size_t)DMODEL * DMODEL;
  __bf16* WVH    = hb + ho; ho += (size_t)DMODEL * DMODEL;
  __bf16* WOH    = hb + ho; ho += (size_t)DMODEL * DMODEL;
  __bf16* WPW1H  = hb + ho; ho += (size_t)512 * DMODEL;
  __bf16* WPW2H  = hb + ho; ho += (size_t)DMODEL * DMODEL;
  __bf16* WPHQH  = hb + ho; ho += (size_t)DMODEL * DMODEL;
  __bf16* WSWKH  = hb + ho; ho += (size_t)DMODEL * DMODEL;
  __bf16* WPHKH  = hb + ho; ho += (size_t)DMODEL * DMODEL;

  float* outp = (float*)d_out;

  // ---- one-time bf16 conversions (inputs + weights) ----
  launch_cvt(stream, model,  MODELH, (size_t)TOKS * DMODEL);
  launch_cvt(stream, ctx,    CTXH,   (size_t)KSWTOK * DMODEL);
  launch_cvt(stream, xph,    XPHH,   (size_t)KPHTOK * DMODEL);
  launch_cvt(stream, ffm_w1, WFFM1H, (size_t)FFDIM * DMODEL);
  launch_cvt(stream, ffm_w2, WFFM2H, (size_t)DMODEL * FFDIM);
  launch_cvt(stream, ff_w1,  WFF1H,  (size_t)FFDIM * DMODEL);
  launch_cvt(stream, ff_w2,  WFF2H,  (size_t)DMODEL * FFDIM);
  launch_cvt(stream, wq,     WQH,    (size_t)DMODEL * DMODEL);
  launch_cvt(stream, wk,     WKH,    (size_t)DMODEL * DMODEL);
  launch_cvt(stream, wv,     WVH,    (size_t)DMODEL * DMODEL);
  launch_cvt(stream, wo,     WOH,    (size_t)DMODEL * DMODEL);
  launch_cvt(stream, pw1_w,  WPW1H,  (size_t)512 * DMODEL);
  launch_cvt(stream, pw2_w,  WPW2H,  (size_t)DMODEL * DMODEL);
  launch_cvt(stream, ph_q_w, WPHQH,  (size_t)DMODEL * DMODEL);
  launch_cvt(stream, sw_k_w, WSWKH,  (size_t)DMODEL * DMODEL);
  launch_cvt(stream, ph_k_w, WPHKH,  (size_t)DMODEL * DMODEL);

  // x = model_embed
  hipMemcpyAsync(X, model, (size_t)TOKS * DMODEL * sizeof(float),
                 hipMemcpyDeviceToDevice, stream);

  // ---- macaron FFN: x += 0.5 * ffn(LN(x)) ----
  layernorm_k<<<TOKS / 8, 256, 0, stream>>>(X, ln_ffm_g, ln_ffm_b, nullptr, LNBH, TOKS);
  launch_gemm(stream, LNBH, 0, 0, DMODEL, WFFM1H, 0, 0, DMODEL, ffm_b1,
              nullptr, 0, 0, 0, FFBH, 0, 0, FFDIM, true,
              TOKS, FFDIM, DMODEL, 1, 1, 1.0f, 1);
  launch_gemm(stream, FFBH, 0, 0, FFDIM, WFFM2H, 0, 0, FFDIM, ffm_b2,
              X, 0, 0, DMODEL, X, 0, 0, DMODEL, false,
              TOKS, DMODEL, FFDIM, 1, 1, 0.5f, 0);

  // ---- MHA: x += attn(LN(x)) ----
  layernorm_k<<<TOKS / 8, 256, 0, stream>>>(X, ln_mha_g, ln_mha_b, nullptr, LNBH, TOKS);
  launch_gemm(stream, LNBH, 0, 0, DMODEL, WQH, 0, 0, DMODEL, wq_b,
              nullptr, 0, 0, 0, QH, 0, 0, DMODEL, true, TOKS, DMODEL, DMODEL, 1, 1, 1.0f, 0);
  launch_gemm(stream, LNBH, 0, 0, DMODEL, WKH, 0, 0, DMODEL, wk_b,
              nullptr, 0, 0, 0, KH, 0, 0, DMODEL, true, TOKS, DMODEL, DMODEL, 1, 1, 1.0f, 0);
  launch_gemm(stream, LNBH, 0, 0, DMODEL, WVH, 0, 0, DMODEL, wv_b,
              nullptr, 0, 0, 0, VH, 0, 0, DMODEL, true, TOKS, DMODEL, DMODEL, 1, 1, 1.0f, 0);
  vtrans_k<<<(TOKS * DMODEL) / 256, 256, 0, stream>>>(VH, VTH);
  // scores[b,h] = Q[b,:,h,:] K[b,:,h,:]^T / 8   -> f32
  launch_gemm(stream, QH, 65536, 64, DMODEL, KH, 65536, 64, DMODEL, nullptr,
              nullptr, 0, 0, 0, ATTN, 262144, 65536, 256, false,
              256, 256, DHEAD, NHEAD, 16, 0.125f, 0);
  softmax256_k<<<16 * 256, 256, 0, stream>>>(ATTN, ATTNH);
  // o[b,:,h,:] = attn[b,h] V[b,:,h,:]   (B from transposed V, K-contiguous)
  launch_gemm(stream, ATTNH, 262144, 65536, 256, VTH, 65536, 16384, 256, nullptr,
              nullptr, 0, 0, 0, OBH, 65536, 64, DMODEL, true,
              256, DHEAD, 256, NHEAD, 16, 1.0f, 0);
  launch_gemm(stream, OBH, 0, 0, DMODEL, WOH, 0, 0, DMODEL, wo_b,
              X, 0, 0, DMODEL, X, 0, 0, DMODEL, false, TOKS, DMODEL, DMODEL, 1, 1, 1.0f, 0);

  // ---- conv module: x += conv(LN(x)) ----
  layernorm_k<<<TOKS / 8, 256, 0, stream>>>(X, ln_conv_g, ln_conv_b, nullptr, LNBH, TOKS);
  launch_gemm(stream, LNBH, 0, 0, DMODEL, WPW1H, 0, 0, DMODEL, pw1_b,
              nullptr, 0, 0, 0, PW1B, 0, 0, 512, false, TOKS, 512, DMODEL, 1, 1, 1.0f, 0);
  glu_k<<<(TOKS * DMODEL) / 256, 256, 0, stream>>>(PW1B, GLUB);
  dwconv_bn_swish_k<<<(TOKS * DMODEL) / 256, 256, 0, stream>>>(GLUB, dw_w, dw_b,
                                                               bn_g, bn_b, CONVH);
  launch_gemm(stream, CONVH, 0, 0, DMODEL, WPW2H, 0, 0, DMODEL, pw2_b,
              X, 0, 0, DMODEL, X, 0, 0, DMODEL, false, TOKS, DMODEL, DMODEL, 1, 1, 1.0f, 0);

  // ---- final FFN: x += 0.5 * ffn(LN(x)) ----
  layernorm_k<<<TOKS / 8, 256, 0, stream>>>(X, ln_ff_g, ln_ff_b, nullptr, LNBH, TOKS);
  launch_gemm(stream, LNBH, 0, 0, DMODEL, WFF1H, 0, 0, DMODEL, ff_b1,
              nullptr, 0, 0, 0, FFBH, 0, 0, FFDIM, true,
              TOKS, FFDIM, DMODEL, 1, 1, 1.0f, 1);
  launch_gemm(stream, FFBH, 0, 0, FFDIM, WFF2H, 0, 0, FFDIM, ff_b2,
              X, 0, 0, DMODEL, X, 0, 0, DMODEL, false,
              TOKS, DMODEL, FFDIM, 1, 1, 0.5f, 0);

  // query_sw2 = LN_final(x) + model_embed  -> bf16
  layernorm_k<<<TOKS / 8, 256, 0, stream>>>(X, ln_fin_g, ln_fin_b, model, QSWH, TOKS);

  // projection heads (GEMM + f32 residual, bf16 out)
  launch_gemm(stream, MODELH, 0, 0, DMODEL, WPHQH, 0, 0, DMODEL, ph_q_b,
              model, 0, 0, DMODEL, QPHH, 0, 0, DMODEL, true,
              TOKS, DMODEL, DMODEL, 1, 1, 1.0f, 0);
  launch_gemm(stream, CTXH, 0, 0, DMODEL, WSWKH, 0, 0, DMODEL, sw_k_b,
              ctx, 0, 0, DMODEL, KEYSWH, 0, 0, DMODEL, true,
              KSWTOK, DMODEL, DMODEL, 1, 1, 1.0f, 0);
  launch_gemm(stream, XPHH, 0, 0, DMODEL, WPHKH, 0, 0, DMODEL, ph_k_b,
              xph, 0, 0, DMODEL, KEYPHH, 0, 0, DMODEL, true,
              KPHTOK, DMODEL, DMODEL, 1, 1, 1.0f, 0);

  // fused maxsim: sw + ph
  {
    dim3 grid(TOKS / 16, NCTX / 8);
    similarity_k<<<grid, 256, 0, stream>>>(QSWH, KEYSWH, QPHH, KEYPHH, MAXS);
  }

  // softmax over contexts -> output
  softmax2000_k<<<TOKS, 256, 0, stream>>>(MAXS, outp);
}
